// PWNet3DH2O_3470333575480
// MI455X (gfx1250) — matrix operand order repaired and checked by reading, not compiled, not run on hardware
//
#include <hip/hip_runtime.h>
#include <hip/hip_bf16.h>

typedef __attribute__((ext_vector_type(16))) _Float16 v16h;
typedef __attribute__((ext_vector_type(8)))  _Float16 v8h;
typedef __attribute__((ext_vector_type(8)))  float    v8f;

#define NGROUP 3
#define CPG    128
#define GCPG   (NGROUP * CPG)   // 384

// ---------------------------------------------------------------------------
// Workspace layout (ints):
//   meta[0..2]  : per-group counts
//   meta[4..6]  : per-group scatter cursors
//   meta[8..11] : per-group block offsets (exclusive prefix of ceil(cnt/256)),
//                 meta[11] = total compute blocks
//   meta[12..14]: per-group element offsets (exclusive prefix of cnt)
//   meta[16 ...]: idx[N] compacted element ids
// ---------------------------------------------------------------------------

__global__ void pw_init(int* meta) {
    if (threadIdx.x < 3) meta[threadIdx.x] = 0;
}

__global__ __launch_bounds__(256) void pw_count(const float* __restrict__ x,
                                                int* __restrict__ meta, int N) {
    __shared__ int sc[3];
    const int tid = threadIdx.x;
    if (tid < 3) sc[tid] = 0;
    __syncthreads();
    const int e = blockIdx.x * 256 + tid;
    if (e < N) {
        const int b = e >> 4, l = e & 15;
        const float x0 = x[b * 48 + l];
        const float x1 = x[b * 48 + 16 + l];
        const int g = (x0 > 1e-6f) ? 0 : ((x1 > 1e-6f) ? 1 : 2);
        atomicAdd(&sc[g], 1);
    }
    __syncthreads();
    if (tid < 3) atomicAdd(&meta[tid], sc[tid]);
}

__global__ void pw_offsets(int* meta) {
    if (threadIdx.x == 0 && blockIdx.x == 0) {
        const int c0 = meta[0], c1 = meta[1], c2 = meta[2];
        // element offsets
        meta[12] = 0; meta[13] = c0; meta[14] = c0 + c1;
        // scatter cursors start at element offsets
        meta[4] = 0; meta[5] = c0; meta[6] = c0 + c1;
        // block offsets (256 elements per compute block)
        const int g0 = (c0 + 255) >> 8, g1 = (c1 + 255) >> 8, g2 = (c2 + 255) >> 8;
        meta[8] = 0; meta[9] = g0; meta[10] = g0 + g1; meta[11] = g0 + g1 + g2;
    }
}

__global__ __launch_bounds__(256) void pw_scatter(const float* __restrict__ x,
                                                  int* __restrict__ meta,
                                                  int* __restrict__ idx, int N) {
    __shared__ int sc[3];
    __shared__ int sbase[3];
    const int tid = threadIdx.x;
    if (tid < 3) sc[tid] = 0;
    __syncthreads();
    const int e = blockIdx.x * 256 + tid;
    int g = 0, local = 0;
    bool ok = (e < N);
    if (ok) {
        const int b = e >> 4, l = e & 15;
        const float x0 = x[b * 48 + l];
        const float x1 = x[b * 48 + 16 + l];
        g = (x0 > 1e-6f) ? 0 : ((x1 > 1e-6f) ? 1 : 2);
        local = atomicAdd(&sc[g], 1);
    }
    __syncthreads();
    if (tid < 3) sbase[tid] = atomicAdd(&meta[4 + tid], sc[tid]);
    __syncthreads();
    if (ok) idx[sbase[g] + local] = e;
}

// ---------------------------------------------------------------------------
// Main compute: one block = 256 elements of ONE bucket (group). 8 waves; each
// wave owns TWO 16-column tiles so every A-fragment (W2 tile, from LDS) feeds
// two v_wmma_f32_16x16x32_f16 ops: D(128x32) = W2[g](128x128) * H1(128x32).
// ---------------------------------------------------------------------------
__global__ __launch_bounds__(256) void pw_compute(
    const float* __restrict__ x,
    const float* __restrict__ W1, const float* __restrict__ bia1,
    const float* __restrict__ W2, const float* __restrict__ bia2,
    const float* __restrict__ Wl, const float* __restrict__ bial,
    const int*  __restrict__ meta, const int* __restrict__ idx,
    float* __restrict__ out)
{
    const int bid = blockIdx.x;
    if (bid >= meta[11]) return;               // block-uniform early exit

    // Which bucket does this block serve?
    int g, chunk;
    if (bid < meta[9])       { g = 0; chunk = bid; }
    else if (bid < meta[10]) { g = 1; chunk = bid - meta[9]; }
    else                     { g = 2; chunk = bid - meta[10]; }
    const int cntg  = meta[g];
    const int ebase = meta[12 + g] + chunk * 256;
    int nvalid = cntg - chunk * 256;
    if (nvalid > 256) nvalid = 256;

    __shared__ __align__(16) _Float16 sW2[CPG * CPG];   // 32 KB, f16
    __shared__ float sW1[CPG], sb1[CPG], sb2[CPG], sWl[3 * CPG], sbl[3];

    const int tid = threadIdx.x;
    // Stage this group's weights (f32 -> f16 for W2).
    const float* w2g = W2 + g * CPG * CPG;
    #pragma unroll 4
    for (int i = tid; i < CPG * CPG; i += 256) sW2[i] = (_Float16)w2g[i];
    for (int i = tid; i < CPG; i += 256) {
        sW1[i] = W1[g * CPG + i];
        sb1[i] = bia1[g * CPG + i];
        sb2[i] = bia2[g * CPG + i];
    }
    for (int i = tid; i < 3 * CPG; i += 256) {
        const int o = i >> 7, c = i & 127;
        sWl[i] = Wl[o * GCPG + g * CPG + c];
    }
    if (tid < 3) sbl[tid] = bial[tid];
    __syncthreads();

    const int wv   = tid >> 5;
    const int lane = tid & 31;
    const int n    = lane & 15;    // tile column
    const int hi   = lane >> 4;    // lane half (wave32)

    // Two column sets per wave.
    const int  eposA  = wv * 32 + n;
    const int  eposB  = eposA + 16;
    const bool validA = eposA < nvalid;
    const bool validB = eposB < nvalid;
    const int  eA = validA ? idx[ebase + eposA] : 0;
    const int  eB = validB ? idx[ebase + eposB] : 0;
    const int  bA = eA >> 4, lA = eA & 15;
    const int  bB = eB >> 4, lB = eB & 15;
    const float rA = validA ? x[bA * 48 + g * 16 + lA] : 0.0f;  // = r_square
    const float rB = validB ? x[bB * 48 + g * 16 + lB] : 0.0f;

    // Build B fragments: H1[k][n] = relu(W1[g][k]*r + b1[g][k]).
    // 16-bit B 32x16 layout: lane half hi holds K runs {hi*8..hi*8+7} and {16+hi*8..16+hi*8+7}.
    v16h bfragA[4], bfragB[4];
    #pragma unroll
    for (int kc = 0; kc < 4; ++kc) {
        #pragma unroll
        for (int j = 0; j < 16; ++j) {
            const int k = kc * 32 + hi * 8 + j + (j >= 8 ? 8 : 0);
            const float w1k = sW1[k], b1k = sb1[k];
            bfragA[kc][j] = (_Float16)fmaxf(w1k * rA + b1k, 0.0f);
            bfragB[kc][j] = (_Float16)fmaxf(w1k * rB + b1k, 0.0f);
        }
    }

    float yA0 = 0.0f, yA1 = 0.0f, yA2 = 0.0f;
    float yB0 = 0.0f, yB1 = 0.0f, yB2 = 0.0f;

    #pragma unroll
    for (int mt = 0; mt < 8; ++mt) {
        v8f accA, accB;
        #pragma unroll
        for (int v = 0; v < 8; ++v) { accA[v] = 0.0f; accB[v] = 0.0f; }

        #pragma unroll
        for (int kc = 0; kc < 4; ++kc) {
            // 16-bit A 16x32 layout: lane row m = lane&15; lane half selects
            // the interleaved K runs {hi*8..hi*8+7} and {16+hi*8..}.
            const _Float16* ap = &sW2[(mt * 16 + n) * CPG + kc * 32 + hi * 8];
            const v8h lo  = *(const v8h*)ap;
            const v8h hi8 = *(const v8h*)(ap + 16);
            const v16h afrag = __builtin_shufflevector(
                lo, hi8, 0, 1, 2, 3, 4, 5, 6, 7, 8, 9, 10, 11, 12, 13, 14, 15);
            accA = __builtin_amdgcn_wmma_f32_16x16x32_f16(
                false, afrag, false, bfragA[kc], (short)0, accA, false, false);
            accB = __builtin_amdgcn_wmma_f32_16x16x32_f16(
                false, afrag, false, bfragB[kc], (short)0, accB, false, false);
        }
        // Epilogue: h2 = relu(acc + b2); fold the 3x128 output contraction.
        // C/D layout: VGPR v holds row m = mt*16 + v + 8*hi, column n.
        #pragma unroll
        for (int v = 0; v < 8; ++v) {
            const int m = mt * 16 + v + 8 * hi;
            const float b2v = sb2[m];
            const float w0 = sWl[0 * CPG + m];
            const float w1 = sWl[1 * CPG + m];
            const float w2 = sWl[2 * CPG + m];
            const float hA = fmaxf(accA[v] + b2v, 0.0f);
            const float hB = fmaxf(accB[v] + b2v, 0.0f);
            yA0 += w0 * hA; yA1 += w1 * hA; yA2 += w2 * hA;
            yB0 += w0 * hB; yB1 += w1 * hB; yB2 += w2 * hB;
        }
    }

    // Each column's partials live in lanes n and n+16 -> reduce across halves.
    const float tA0 = yA0 + __shfl_xor(yA0, 16);
    const float tA1 = yA1 + __shfl_xor(yA1, 16);
    const float tA2 = yA2 + __shfl_xor(yA2, 16);
    const float tB0 = yB0 + __shfl_xor(yB0, 16);
    const float tB1 = yB1 + __shfl_xor(yB1, 16);
    const float tB2 = yB2 + __shfl_xor(yB2, 16);

    if (hi == 0) {
        if (validA) {
            const float w = 1.0f / (rA * rA * rA + 0.1f);
            out[bA * 48 +  0 + lA] = tanhf(tA0 + sbl[0]) * w;
            out[bA * 48 + 16 + lA] = tanhf(tA1 + sbl[1]) * w;
            out[bA * 48 + 32 + lA] = tanhf(tA2 + sbl[2]) * w;
        }
        if (validB) {
            const float w = 1.0f / (rB * rB * rB + 0.1f);
            out[bB * 48 +  0 + lB] = tanhf(tB0 + sbl[0]) * w;
            out[bB * 48 + 16 + lB] = tanhf(tB1 + sbl[1]) * w;
            out[bB * 48 + 32 + lB] = tanhf(tB2 + sbl[2]) * w;
        }
    }
}

// ---------------------------------------------------------------------------
extern "C" void kernel_launch(void* const* d_in, const int* in_sizes, int n_in,
                              void* d_out, int out_size, void* d_ws, size_t ws_size,
                              hipStream_t stream) {
    const float* x   = (const float*)d_in[0];
    const float* W1  = (const float*)d_in[1];
    const float* b1  = (const float*)d_in[2];
    const float* W2  = (const float*)d_in[3];
    const float* b2  = (const float*)d_in[4];
    const float* Wl  = (const float*)d_in[5];
    const float* bl  = (const float*)d_in[6];
    float* out = (float*)d_out;

    const int N = in_sizes[0] / NGROUP;     // B * L = 262144 element sites
    int* meta = (int*)d_ws;
    int* idx  = meta + 16;

    pw_init<<<1, 32, 0, stream>>>(meta);
    pw_count<<<(N + 255) / 256, 256, 0, stream>>>(x, meta, N);
    pw_offsets<<<1, 32, 0, stream>>>(meta);
    pw_scatter<<<(N + 255) / 256, 256, 0, stream>>>(x, meta, idx, N);
    pw_compute<<<N / 256 + 3, 256, 0, stream>>>(x, W1, b1, W2, b2, Wl, bl,
                                                meta, idx, out);
}
